// SPINN_29497835389443
// MI455X (gfx1250) — matrix operand order repaired and checked by reading, not compiled, not run on hardware
//
#include <hip/hip_runtime.h>
#include <hip/hip_bf16.h>

typedef __attribute__((ext_vector_type(16))) __bf16 v16bf;
typedef __attribute__((ext_vector_type(8)))  float  v8f;

#define B_   64
#define L_   256
#define E_   1024
#define H_   1024
#define T_   511        // 2L-1
#define S_   257        // L+1
#define EPS_ 1e-5f

#define TILE_ELEMS 512  // 32 lanes * 16 bf16 per WMMA operand tile

#define XBYTES   (B_ * 2 * H_ * 2)            // packed X: 64 x 2048 bf16 = 256 KB
#define GATE_BYTES (5 * B_ * 16 * 4)          // 5 gates x 64 rows x 16 cols f32 = 20 KB
#define SMEM_BYTES (XBYTES + GATE_BYTES)      // 276 KB of the WGP's 320 KB LDS

// ---------------- WMMA helpers ----------------

__device__ __forceinline__ v16bf load_tile(const __bf16* __restrict__ base, int tileIdx, int lane) {
    return *(const v16bf*)(base + (size_t)tileIdx * TILE_ELEMS + lane * 16);
}

__device__ __forceinline__ v8f wmma_bf16(v16bf a, v16bf b, v8f c) {
    return __builtin_amdgcn_wmma_f32_16x16x32_bf16(false, a, false, b, (short)0, c, false, false);
}

__device__ __forceinline__ float sigf(float x) { return 1.0f / (1.0f + __expf(-x)); }

// CDNA5 async copy: 16 bytes per lane, global -> LDS, tracked by ASYNCcnt.
__device__ __forceinline__ void async_load_b128(unsigned lds_off, const void* gptr) {
    asm volatile("global_load_async_to_lds_b128 %0, %1, off"
                 :: "v"(lds_off), "v"(gptr)
                 : "memory");
}
__device__ __forceinline__ void wait_asynccnt0() {
    asm volatile("s_wait_asynccnt 0x0" ::: "memory");
}

// ---------------- Packing kernels ----------------
// A-tile layout (16-bit A 16x32, ISA 7.12.2): lane = m%16 + 16*((k>>3)&1),
//   elem e: k = (e/8)*16 + (lane/16)*8 + (e%8)
__global__ void pack_a_kernel(const float* __restrict__ in, __bf16* __restrict__ out, int M, int K) {
    int idx = blockIdx.x * blockDim.x + threadIdx.x;
    if (idx >= M * K) return;
    int e    = idx & 15;
    int lane = (idx >> 4) & 31;
    int tile = idx >> 9;
    int Ktiles = K >> 5;
    int mt = tile / Ktiles, kt = tile - mt * Ktiles;
    int m = mt * 16 + (lane & 15);
    int k = kt * 32 + ((e >= 8) ? 16 : 0) + ((lane >= 16) ? 8 : 0) + (e & 7);
    out[idx] = (__bf16)in[(size_t)m * K + k];
}

// B-tile layout (16-bit B KxN): lane holds column n = n0 + lane%16,
//   elem e: k = k0 + (e/8)*16 + (lane/16)*8 + (e%8)
__global__ void pack_b_kernel(const float* __restrict__ in, __bf16* __restrict__ out, int K, int N) {
    int idx = blockIdx.x * blockDim.x + threadIdx.x;
    if (idx >= K * N) return;
    int e    = idx & 15;
    int lane = (idx >> 4) & 31;
    int tile = idx >> 9;
    int Ktiles = K >> 5;
    int nt = tile / Ktiles, kt = tile - nt * Ktiles;
    int n = nt * 16 + (lane & 15);
    int k = kt * 32 + ((e >= 8) ? 16 : 0) + ((lane >= 16) ? 8 : 0) + (e & 7);
    out[idx] = (__bf16)in[(size_t)k * N + n];
}

// ---------------- Word projection: pass 1 (stats) ----------------
__global__ void proj_pass1(const __bf16* __restrict__ Apack, const __bf16* __restrict__ Bpack,
                           const float* __restrict__ bias,
                           float* __restrict__ P1, float* __restrict__ P2) {
    int lane = threadIdx.x & 31;
    int wave = threadIdx.x >> 5;
    int mt = blockIdx.x;
    int nt = blockIdx.y * 8 + wave;
    const int KT = E_ / 32;
    v8f acc = {0.f,0.f,0.f,0.f,0.f,0.f,0.f,0.f};
    for (int kt = 0; kt < KT; ++kt) {
        v16bf a = load_tile(Apack, mt * KT + kt, lane);
        v16bf b = load_tile(Bpack, nt * KT + kt, lane);
        acc = wmma_bf16(a, b, acc);
    }
    int n = nt * 16 + (lane & 15);
    float bv = bias[n];
    float s1 = 0.f, s2 = 0.f;
#pragma unroll
    for (int r = 0; r < 8; ++r) { float v = acc[r] + bv; s1 += v; s2 += v * v; }
    s1 += __shfl_xor(s1, 16, 32);
    s2 += __shfl_xor(s2, 16, 32);
    if (lane < 16) {
        P1[(size_t)mt * (2 * H_) + n] = s1;
        P2[(size_t)mt * (2 * H_) + n] = s2;
    }
}

__global__ void bn_finalize(const float* __restrict__ P1, const float* __restrict__ P2,
                            const float* __restrict__ gamma, const float* __restrict__ beta,
                            const float* __restrict__ b_word,
                            float* __restrict__ sc, float* __restrict__ sh) {
    int n = blockIdx.x * blockDim.x + threadIdx.x;
    if (n >= 2 * H_) return;
    float s1 = 0.f, s2 = 0.f;
    for (int mt = 0; mt < (B_ * L_) / 16; ++mt) {   // fixed order -> deterministic
        s1 += P1[(size_t)mt * (2 * H_) + n];
        s2 += P2[(size_t)mt * (2 * H_) + n];
    }
    float N = (float)(B_ * L_);
    float mean = s1 / N;
    float var  = s2 / N - mean * mean;
    float s = gamma[n] * rsqrtf(var + EPS_);
    sc[n] = s;
    sh[n] = beta[n] + (b_word[n] - mean) * s;
}

// ---------------- Word projection: pass 2 (normalize + split) ----------------
__global__ void proj_pass2(const __bf16* __restrict__ Apack, const __bf16* __restrict__ Bpack,
                           const float* __restrict__ sc, const float* __restrict__ sh,
                           float* __restrict__ h_sent, float* __restrict__ c_sent) {
    int lane = threadIdx.x & 31;
    int wave = threadIdx.x >> 5;
    int mt = blockIdx.x;
    int nt = blockIdx.y * 8 + wave;
    const int KT = E_ / 32;
    v8f acc = {0.f,0.f,0.f,0.f,0.f,0.f,0.f,0.f};
    for (int kt = 0; kt < KT; ++kt) {
        v16bf a = load_tile(Apack, mt * KT + kt, lane);
        v16bf b = load_tile(Bpack, nt * KT + kt, lane);
        acc = wmma_bf16(a, b, acc);
    }
    int n = nt * 16 + (lane & 15);
    float s = sc[n], t = sh[n];
#pragma unroll
    for (int r = 0; r < 8; ++r) {
        int m = mt * 16 + r + ((lane >= 16) ? 8 : 0);
        float v = acc[r] * s + t;
        if (n < H_) h_sent[(size_t)m * H_ + n] = v;
        else        c_sent[(size_t)m * H_ + (n - H_)] = v;
    }
}

// ---------------- Schedule precompute (depends only on transitions) ----------------
__global__ void schedule_kernel(const int* __restrict__ transitions, const int* __restrict__ num_ops,
                                int* __restrict__ act_a, int* __restrict__ il_a, int* __restrict__ ir_a,
                                int* __restrict__ widx_a, int* __restrict__ bpc_a,
                                int* __restrict__ any_shift, int* __restrict__ any_reduce,
                                int* __restrict__ sp_final) {
    int b = threadIdx.x;
    if (b >= B_) return;
    int sp = 0, bp = 0;
    for (int t = 0; t < T_; ++t) {
        int act = transitions[b * T_ + t];
        if (t >= num_ops[b]) act = 0;
        int is_s = (act == 1), is_r = (act == 2);
        int ir  = min(max(sp - 1, 0), S_ - 1);
        int il  = min(max(sp - 2, 0), S_ - 1);
        int bpc = min(max(bp, 0), L_ - 1);
        int widx = is_s ? min(max(sp, 0), S_ - 1) : il;
        act_a[t * B_ + b] = act;  il_a[t * B_ + b] = il;  ir_a[t * B_ + b] = ir;
        widx_a[t * B_ + b] = widx; bpc_a[t * B_ + b] = bpc;
        if (is_s) atomicOr(&any_shift[t], 1);
        if (is_r) atomicOr(&any_reduce[t], 1);
        sp += is_s - is_r;
        bp += is_s;
    }
    sp_final[b] = sp;
}

// ---------------- Per-step: shift pushes + gather A operand ----------------
// idx over B*2H. For k<H also performs the SHIFT push (slot sp), which is disjoint
// from the gathered slots (sp-2, sp-1) within a row, so a single pass is safe.
__global__ void step_gather(int t,
                            const int* __restrict__ act_a, const int* __restrict__ il_a,
                            const int* __restrict__ ir_a, const int* __restrict__ widx_a,
                            const int* __restrict__ bpc_a, const int* __restrict__ any_reduce,
                            const float* __restrict__ h_sent, const float* __restrict__ c_sent,
                            float* __restrict__ stack_h, float* __restrict__ stack_c,
                            __bf16* __restrict__ Xpack) {
    int idx = blockIdx.x * blockDim.x + threadIdx.x;
    if (idx >= B_ * 2 * H_) return;
    int b = idx >> 11;
    int k = idx & (2 * H_ - 1);
    int act = act_a[t * B_ + b];
    if (k < H_ && act == 1) {
        int widx = widx_a[t * B_ + b], bpc = bpc_a[t * B_ + b];
        stack_h[((size_t)b * S_ + widx) * H_ + k] = h_sent[((size_t)b * L_ + bpc) * H_ + k];
        stack_c[((size_t)b * S_ + widx) * H_ + k] = c_sent[((size_t)b * L_ + bpc) * H_ + k];
    }
    if (any_reduce[t]) {
        int j, slot;
        if (k < H_) { j = k;       slot = il_a[t * B_ + b]; }
        else        { j = k - H_;  slot = ir_a[t * B_ + b]; }
        float v = stack_h[((size_t)b * S_ + slot) * H_ + j];
        int mt = b >> 4, m = b & 15;
        int kt = k >> 5, kk = k & 31;
        int lane = m + (((kk >> 3) & 1) << 4);
        int e = ((kk >> 4) << 3) + (kk & 7);
        const int KT2 = (2 * H_) / 32;
        Xpack[(((size_t)mt * KT2 + kt) * 32 + lane) * 16 + e] = (__bf16)v;
    }
}

// ---------------- Per-step: fused 5-gate GEMM + TreeLSTM cell ----------------
// grid 64 (j-tiles), block 160 = 5 waves, one wave per gate so every W_reduce
// tile is loaded exactly once per step (L2-resident, roofline minimum).
// Packed X (256 KB) is staged into LDS once per block via the CDNA5 async
// global->LDS path; WMMA A operands then come from ds_load. Gate tiles are
// exchanged through a 20 KB LDS buffer for the cell update.
__global__ void __launch_bounds__(160)
step_reduce2(int t,
             const int* __restrict__ act_a, const int* __restrict__ il_a,
             const int* __restrict__ ir_a, const int* __restrict__ any_reduce,
             const __bf16* __restrict__ Xpack, const __bf16* __restrict__ Wrpack,
             const float* __restrict__ b_reduce,
             float* __restrict__ stack_h, float* __restrict__ stack_c) {
    extern __shared__ char smem[];                 // [XBYTES) X stage, [XBYTES..) gates
    if (!any_reduce[t]) return;
    const int tid  = threadIdx.x;
    const int lane = tid & 31;
    const int g    = tid >> 5;                     // wave id == gate id, 0..4
    const int jt   = blockIdx.x;                   // 0..63
    const int KT2  = (2 * H_) / 32;                // 64

    // stage packed X into LDS (16384 x 16B transfers across 160 lanes)
    {
        const char* src = (const char*)Xpack;
        unsigned lds_base = (unsigned)(unsigned long long)(size_t)smem; // low 32 bits = LDS offset
        for (int i = tid; i < XBYTES / 16; i += 160) {
            async_load_b128(lds_base + (unsigned)(i * 16), src + (size_t)i * 16);
        }
        wait_asynccnt0();          // this wave's copies complete
    }
    __syncthreads();               // all waves' copies visible

    v8f acc[4];
#pragma unroll
    for (int mt = 0; mt < 4; ++mt) acc[mt] = (v8f){0.f,0.f,0.f,0.f,0.f,0.f,0.f,0.f};

    const __bf16* Xlds = (const __bf16*)smem;
    const int nt = g * (H_ / 16) + jt;             // gate g, j-tile jt
    for (int kt = 0; kt < KT2; ++kt) {
        v16bf b = load_tile(Wrpack, nt * KT2 + kt, lane);   // unique per wave
#pragma unroll
        for (int mt = 0; mt < 4; ++mt) {
            v16bf a = *(const v16bf*)(Xlds + ((size_t)(mt * KT2 + kt)) * TILE_ELEMS + lane * 16);
            acc[mt] = wmma_bf16(a, b, acc[mt]);
        }
    }

    // publish gate tiles: gates[g][m 0..63][nl 0..15]
    float* gmem = (float*)(smem + XBYTES);
    {
        int nl = lane & 15;
#pragma unroll
        for (int mt = 0; mt < 4; ++mt)
#pragma unroll
            for (int r = 0; r < 8; ++r) {
                int m = mt * 16 + r + ((lane >= 16) ? 8 : 0);
                gmem[(g * B_ + m) * 16 + nl] = acc[mt][r];
            }
    }
    __syncthreads();

    // TreeLSTM cell update + masked stack writes for this block's 16 columns
    const int j0 = jt * 16;
    for (int c = tid; c < B_ * 16; c += 160) {
        int b  = c >> 4;
        int nl = c & 15;
        if (act_a[t * B_ + b] != 2) continue;
        int j  = j0 + nl;
        int il = il_a[t * B_ + b], ir = ir_a[t * B_ + b];
        float ig  = sigf (gmem[(0 * B_ + b) * 16 + nl] + b_reduce[0 * H_ + j]);
        float flg = sigf (gmem[(1 * B_ + b) * 16 + nl] + b_reduce[1 * H_ + j]);
        float frg = sigf (gmem[(2 * B_ + b) * 16 + nl] + b_reduce[2 * H_ + j]);
        float og  = sigf (gmem[(3 * B_ + b) * 16 + nl] + b_reduce[3 * H_ + j]);
        float ug  = tanhf(gmem[(4 * B_ + b) * 16 + nl] + b_reduce[4 * H_ + j]);
        float c_l = stack_c[((size_t)b * S_ + il) * H_ + j];
        float c_r = stack_c[((size_t)b * S_ + ir) * H_ + j];
        float c_new = flg * c_l + frg * c_r + ig * ug;
        float h_new = og * tanhf(c_new);
        stack_c[((size_t)b * S_ + il) * H_ + j] = c_new;
        stack_h[((size_t)b * S_ + il) * H_ + j] = h_new;
    }
}

__global__ void final_kernel(const int* __restrict__ sp_final, const float* __restrict__ stack_h,
                             float* __restrict__ out) {
    int idx = blockIdx.x * blockDim.x + threadIdx.x;
    if (idx >= B_ * H_) return;
    int b = idx >> 10, j = idx & (H_ - 1);
    int slot = min(max(sp_final[b] - 1, 0), S_ - 1);
    out[idx] = stack_h[((size_t)b * S_ + slot) * H_ + j];
}

// ---------------- Host launch ----------------
extern "C" void kernel_launch(void* const* d_in, const int* in_sizes, int n_in,
                              void* d_out, int out_size, void* d_ws, size_t ws_size,
                              hipStream_t stream) {
    const float* sentence    = (const float*)d_in[0];
    const int*   transitions = (const int*)d_in[1];
    const int*   num_ops     = (const int*)d_in[2];
    const float* W_word      = (const float*)d_in[3];
    const float* b_word      = (const float*)d_in[4];
    const float* gamma       = (const float*)d_in[5];
    const float* beta        = (const float*)d_in[6];
    const float* W_reduce    = (const float*)d_in[7];
    const float* b_reduce    = (const float*)d_in[8];

    char* w = (char*)d_ws;
    size_t off = 0;
    auto alloc = [&](size_t bytes) -> void* {
        void* p = w + off;
        off = (off + bytes + 255) & ~(size_t)255;
        return p;
    };
    const int M = B_ * L_;                 // 16384
    __bf16* Asent  = (__bf16*)alloc((size_t)M * E_ * 2);              // packed sentence (bf16)
    __bf16* Wwp    = (__bf16*)alloc((size_t)E_ * 2 * H_ * 2);         // packed W_word
    __bf16* Wrp    = (__bf16*)alloc((size_t)(2 * H_) * (5 * H_) * 2); // packed W_reduce
    float*  h_sent = (float*)alloc((size_t)M * H_ * 4);
    float*  c_sent = (float*)alloc((size_t)M * H_ * 4);
    float*  stk_h  = (float*)alloc((size_t)B_ * S_ * H_ * 4);
    float*  stk_c  = (float*)alloc((size_t)B_ * S_ * H_ * 4);
    __bf16* Xp     = (__bf16*)alloc((size_t)B_ * 2 * H_ * 2);
    float*  P1     = (float*)alloc((size_t)(M / 16) * 2 * H_ * 4);
    float*  P2     = (float*)alloc((size_t)(M / 16) * 2 * H_ * 4);
    float*  sc     = (float*)alloc(2 * H_ * 4);
    float*  sh     = (float*)alloc(2 * H_ * 4);
    int* act_a  = (int*)alloc((size_t)T_ * B_ * 4);
    int* il_a   = (int*)alloc((size_t)T_ * B_ * 4);
    int* ir_a   = (int*)alloc((size_t)T_ * B_ * 4);
    int* widx_a = (int*)alloc((size_t)T_ * B_ * 4);
    int* bpc_a  = (int*)alloc((size_t)T_ * B_ * 4);
    int* any_s  = (int*)alloc((size_t)T_ * 4);
    int* any_r  = (int*)alloc((size_t)T_ * 4);
    int* sp_fin = (int*)alloc((size_t)B_ * 4);

    // zero-init state (reference zeros stacks; flags must start 0 each call)
    hipMemsetAsync(stk_h, 0, (size_t)B_ * S_ * H_ * 4, stream);
    hipMemsetAsync(stk_c, 0, (size_t)B_ * S_ * H_ * 4, stream);
    hipMemsetAsync(any_s, 0, (size_t)T_ * 4, stream);
    hipMemsetAsync(any_r, 0, (size_t)T_ * 4, stream);

    // pack operands into WMMA tile layouts
    pack_a_kernel<<<(M * E_ + 255) / 256, 256, 0, stream>>>(sentence, Asent, M, E_);
    pack_b_kernel<<<(E_ * 2 * H_ + 255) / 256, 256, 0, stream>>>(W_word, Wwp, E_, 2 * H_);
    pack_b_kernel<<<((2 * H_) * (5 * H_) + 255) / 256, 256, 0, stream>>>(W_reduce, Wrp, 2 * H_, 5 * H_);

    // projection + batchnorm
    proj_pass1<<<dim3(M / 16, 16), 256, 0, stream>>>(Asent, Wwp, b_word, P1, P2);
    bn_finalize<<<(2 * H_ + 255) / 256, 256, 0, stream>>>(P1, P2, gamma, beta, b_word, sc, sh);
    proj_pass2<<<dim3(M / 16, 16), 256, 0, stream>>>(Asent, Wwp, sc, sh, h_sent, c_sent);

    // schedule
    schedule_kernel<<<1, 64, 0, stream>>>(transitions, num_ops, act_a, il_a, ir_a,
                                          widx_a, bpc_a, any_s, any_r, sp_fin);

    // sequential shift-reduce scan
    for (int t = 0; t < T_; ++t) {
        step_gather<<<(B_ * 2 * H_ + 255) / 256, 256, 0, stream>>>(
            t, act_a, il_a, ir_a, widx_a, bpc_a, any_r, h_sent, c_sent, stk_h, stk_c, Xp);
        step_reduce2<<<H_ / 16, 160, SMEM_BYTES, stream>>>(
            t, act_a, il_a, ir_a, any_r, Xp, Wrp, b_reduce, stk_h, stk_c);
    }

    final_kernel<<<(B_ * H_ + 255) / 256, 256, 0, stream>>>(sp_fin, stk_h, (float*)d_out);
}
